// Attention_87557203296804
// MI455X (gfx1250) — compile-verified, matrix operand order checked by reading, not run on hardware
//
#include <hip/hip_runtime.h>

// ---------------------------------------------------------------------------
// Types for CDNA5 WMMA (wave32): V_WMMA_F32_16X16X32_BF16
// ---------------------------------------------------------------------------
typedef __attribute__((ext_vector_type(16))) __bf16 v16bf;
typedef __attribute__((ext_vector_type(8)))  __bf16 v8bf;
typedef __attribute__((ext_vector_type(8)))  float  v8f;
typedef __attribute__((ext_vector_type(4)))  int    v4i;

#define SHUF16(lo, hi) \
  __builtin_shufflevector((lo), (hi), 0,1,2,3,4,5,6,7,8,9,10,11,12,13,14,15)

__device__ __forceinline__ v8f wmma_bf16(v16bf a, v16bf b, v8f c) {
  return __builtin_amdgcn_wmma_f32_16x16x32_bf16(
      /*neg_a=*/false, a, /*neg_b=*/false, b,
      /*c_mod=*/(short)0, c, /*reuse_a=*/false, /*reuse_b=*/false);
}

__device__ __forceinline__ v8f v8f_zero() {
  v8f z;
#pragma unroll
  for (int i = 0; i < 8; ++i) z[i] = 0.0f;
  return z;
}

// float -> bf16 round-to-nearest-even (bit twiddle).
__device__ __forceinline__ __bf16 f2bf(float f) {
  union { float f; unsigned u; } v; v.f = f;
  unsigned r = v.u + 0x7FFFu + ((v.u >> 16) & 1u);
  union { unsigned short s; __bf16 b; } o;
  o.s = (unsigned short)(r >> 16);
  return o.b;
}

// ---------------------------------------------------------------------------
// Async global -> LDS staging (CDNA5: GLOBAL_LOAD_ASYNC_TO_LDS_B128, ASYNCcnt)
// Builtin parameter types (from hipcc diagnostic): arg0 = int4 AS(1)* (global
// source, prints as "__device__"), arg1 = int4 AS(3)* (LDS dest), then
// imm offset, imm cpol.
// ---------------------------------------------------------------------------
#if defined(__has_builtin)
#  if __has_builtin(__builtin_amdgcn_global_load_async_to_lds_b128)
#    define HAVE_ASYNC_B128 1
#  else
#    define HAVE_ASYNC_B128 0
#  endif
#  if __has_builtin(__builtin_amdgcn_s_wait_asynccnt)
#    define HAVE_WAIT_ASYNC 1
#  else
#    define HAVE_WAIT_ASYNC 0
#  endif
#else
#  define HAVE_ASYNC_B128 0
#  define HAVE_WAIT_ASYNC 0
#endif

__device__ __forceinline__ void copy16_async(void* lds, const void* g) {
#if HAVE_ASYNC_B128
  typedef __attribute__((address_space(1))) v4i* gp_t;
  typedef __attribute__((address_space(3))) v4i* lp_t;
  __builtin_amdgcn_global_load_async_to_lds_b128(
      (gp_t)g, (lp_t)lds, /*offset=*/0, /*cpol=*/0);
#else
  *(uint4*)lds = *(const uint4*)g;
#endif
}

__device__ __forceinline__ void wait_async_all() {
#if HAVE_ASYNC_B128
#  if HAVE_WAIT_ASYNC
  __builtin_amdgcn_s_wait_asynccnt(0);
#  else
  asm volatile("s_wait_asynccnt 0x0" ::: "memory");
#  endif
#endif
}

// ---------------------------------------------------------------------------
// Problem constants
// ---------------------------------------------------------------------------
#define BATCH   4
#define SEQ     2048
#define EMBED   768
#define HEADS   12
#define HD      64
#define NQKV    2304               // 3 * EMBED
#define MTOK    (BATCH * SEQ)      // 8192
#define SCALE   0.125f             // HD^-0.5

// ---------------------------------------------------------------------------
// Elementwise fp32 -> bf16 convert
// ---------------------------------------------------------------------------
__global__ __launch_bounds__(256) void cvt_bf16_kernel(
    const float* __restrict__ in, __bf16* __restrict__ out, int n) {
  int i = blockIdx.x * 256 + threadIdx.x;
  if (i < n) out[i] = f2bf(in[i]);
}

// out[n*rows + k] = bf16(in[k*cols + n])   (transpose K-major -> N-major)
__global__ __launch_bounds__(256) void cvt_transpose_kernel(
    const float* __restrict__ in, __bf16* __restrict__ out, int rows, int cols) {
  int idx = blockIdx.x * 256 + threadIdx.x;
  if (idx >= rows * cols) return;
  int k = idx / cols;
  int n = idx - k * cols;
  out[(size_t)n * rows + k] = f2bf(in[idx]);
}

// ---------------------------------------------------------------------------
// GEMM: 128x128 block, K-step 32, 8 waves (4 in M x 2 in N), wave tile 32x64
// (2x4 WMMA). Double-buffered LDS with async direct-to-LDS prefetch: one
// barrier per K-step, next tile's loads overlap the current tile's WMMAs.
// ---------------------------------------------------------------------------
#define BM 128
#define BN 128
#define BK 32
#define LDS_STRIDE 40   // 32 + 8 pad (bf16 elems); 80B rows, 16B aligned
#define TILE_ELEMS (BM * LDS_STRIDE)

// D = A(bf16, MxK rowmajor) * Bt(bf16, NxK rowmajor) + bias, K = 768.
// MODE 0: scatter into Q/K/Vt with qkv routing.  MODE 1: f32 out + bias.
template <int MODE>
__global__ __launch_bounds__(256) void gemm_kernel(
    const __bf16* __restrict__ A, const __bf16* __restrict__ Bt,
    const float* __restrict__ bias,
    __bf16* __restrict__ Qo, __bf16* __restrict__ Ko, __bf16* __restrict__ Vto,
    float* __restrict__ Fo) {
  __shared__ __bf16 lds_a[2][TILE_ELEMS];
  __shared__ __bf16 lds_b[2][TILE_ELEMS];

  const int tid   = threadIdx.x;
  const int wid   = tid >> 5;
  const int lane  = tid & 31;
  const int l16   = lane & 15;
  const int lhalf = lane >> 4;        // 0 or 1
  const int c0    = lhalf * 8;        // per-lane K sub-offset in operand layout

  const int m_base = blockIdx.x * BM;
  const int n_base = blockIdx.y * BN;
  const int m_off  = (wid >> 1) * 32; // wave M offset: 0,32,64,96
  const int n_off  = (wid & 1) * 64;  // wave N offset: 0,64

  v8f acc[2][4];
#pragma unroll
  for (int i = 0; i < 2; ++i)
#pragma unroll
    for (int j = 0; j < 4; ++j) acc[i][j] = v8f_zero();

  // staging: each thread copies 16 contiguous bf16 (2 x B128) of A and of B
  const int s_row = tid >> 1;          // 0..127
  const int s_col = (tid & 1) * 16;    // 0 or 16
  const __bf16* gA = A  + (size_t)(m_base + s_row) * EMBED + s_col;
  const __bf16* gB = Bt + (size_t)(n_base + s_row) * EMBED + s_col;
  const int s_dst = s_row * LDS_STRIDE + s_col;

  // prologue: prefetch K-tile 0 into buffer 0
  copy16_async(&lds_a[0][s_dst],     gA);
  copy16_async(&lds_a[0][s_dst + 8], gA + 8);
  copy16_async(&lds_b[0][s_dst],     gB);
  copy16_async(&lds_b[0][s_dst + 8], gB + 8);

  int buf = 0;
  for (int k0 = 0; k0 < EMBED; k0 += BK) {
    wait_async_all();     // own prefetch landed
    __syncthreads();      // => everyone's prefetch landed; buf readable

    if (k0 + BK < EMBED) {
      const int nb = buf ^ 1;
      const __bf16* ga = gA + k0 + BK;
      const __bf16* gb = gB + k0 + BK;
      copy16_async(&lds_a[nb][s_dst],     ga);
      copy16_async(&lds_a[nb][s_dst + 8], ga + 8);
      copy16_async(&lds_b[nb][s_dst],     gb);
      copy16_async(&lds_b[nb][s_dst + 8], gb + 8);
    }

    v16bf aop[2], bop[4];
#pragma unroll
    for (int i = 0; i < 2; ++i) {
      const __bf16* p = &lds_a[buf][(m_off + i * 16 + l16) * LDS_STRIDE + c0];
      v8bf lo = *(const v8bf*)p;
      v8bf hi = *(const v8bf*)(p + 16);
      aop[i] = SHUF16(lo, hi);
    }
#pragma unroll
    for (int j = 0; j < 4; ++j) {
      const __bf16* p = &lds_b[buf][(n_off + j * 16 + l16) * LDS_STRIDE + c0];
      v8bf lo = *(const v8bf*)p;
      v8bf hi = *(const v8bf*)(p + 16);
      bop[j] = SHUF16(lo, hi);
    }
#pragma unroll
    for (int i = 0; i < 2; ++i)
#pragma unroll
      for (int j = 0; j < 4; ++j)
        acc[i][j] = wmma_bf16(aop[i], bop[j], acc[i][j]);

    buf ^= 1;
  }

  // Epilogue. C-layout: acc element e of lane = row (lhalf*8+e), col l16.
#pragma unroll
  for (int j = 0; j < 4; ++j) {
    const int n = n_base + n_off + j * 16 + l16;
    const float bv = bias[n];
    if (MODE == 0) {
      const int which = n / EMBED;         // uniform per 16-wide tile
      const int r = n - which * EMBED;
      const int h = r >> 6;
      const int d = r & 63;
#pragma unroll
      for (int i = 0; i < 2; ++i) {
        const int mrow0 = m_base + m_off + i * 16 + lhalf * 8;
#pragma unroll
        for (int e = 0; e < 8; ++e) {
          const int mt = mrow0 + e;
          const int bb = mt >> 11;         // / SEQ
          const int t  = mt & (SEQ - 1);
          const __bf16 hv = f2bf(acc[i][j][e] + bv);
          const size_t bh = (size_t)(bb * HEADS + h);
          if (which == 0)      Qo[(bh * SEQ + t) * HD + d] = hv;
          else if (which == 1) Ko[(bh * SEQ + t) * HD + d] = hv;
          else                 Vto[(bh * HD + d) * (size_t)SEQ + t] = hv;
        }
      }
    } else {
#pragma unroll
      for (int i = 0; i < 2; ++i) {
        const int mrow0 = m_base + m_off + i * 16 + lhalf * 8;
#pragma unroll
        for (int e = 0; e < 8; ++e)
          Fo[(size_t)(mrow0 + e) * EMBED + n] = acc[i][j][e] + bv;
      }
    }
  }
}

// ---------------------------------------------------------------------------
// Fused flash attention: block = (b,h) x 128 queries, 8 waves x 16 queries.
// Key loop tiles of 128 keys staged in LDS (async direct-to-LDS). Q kept in
// registers (A-layout). K-tile LDS region is reused as per-wave P scratch.
// ---------------------------------------------------------------------------
#define KSTR 72    // K-tile LDS row stride (64+8 bf16) = 144B
#define VSTR 136   // V-tile / P-tile LDS row stride (128+8 bf16) = 272B

__global__ __launch_bounds__(256) void attn_kernel(
    const __bf16* __restrict__ Q, const __bf16* __restrict__ K,
    const __bf16* __restrict__ Vt, __bf16* __restrict__ AO) {
  __shared__ __bf16 lds_v[64 * VSTR];        // 17408 B
  __shared__ __bf16 lds_kp[8 * 16 * VSTR];   // 34816 B: K tile, then per-wave P

  const int tid   = threadIdx.x;
  const int wid   = tid >> 5;
  const int lane  = tid & 31;
  const int l16   = lane & 15;
  const int lhalf = lane >> 4;
  const int c0    = lhalf * 8;

  const int bh = blockIdx.x;                 // b*HEADS + h
  const int q0 = blockIdx.y * 128 + wid * 16;

  const __bf16* Qbh = Q  + (size_t)bh * SEQ * HD;
  const __bf16* Kbh = K  + (size_t)bh * SEQ * HD;
  const __bf16* Vbh = Vt + (size_t)bh * HD * SEQ;

  // Q rows in A-operand layout, held in registers (hd=64 -> 2 K-chunks of 32)
  const __bf16* qrow = Qbh + (size_t)(q0 + l16) * HD;
  v8bf q00 = *(const v8bf*)(qrow + c0);
  v8bf q01 = *(const v8bf*)(qrow + 16 + c0);
  v8bf q10 = *(const v8bf*)(qrow + 32 + c0);
  v8bf q11 = *(const v8bf*)(qrow + 48 + c0);
  v16bf qa0 = SHUF16(q00, q01);
  v16bf qa1 = SHUF16(q10, q11);

  float m_i[8], l_i[8];
#pragma unroll
  for (int j = 0; j < 8; ++j) { m_i[j] = -1e30f; l_i[j] = 0.0f; }
  v8f o[4];
#pragma unroll
  for (int t = 0; t < 4; ++t) o[t] = v8f_zero();

  __bf16* my_p = &lds_kp[wid * 16 * VSTR];

  // staging addresses (per thread): K tile 128x64, V tile 64x128
  const int krr = tid >> 1;             // 0..127
  const int kcc = (tid & 1) * 32;       // 0 or 32
  const int vrr = tid >> 2;             // 0..63
  const int vcc = (tid & 3) * 32;       // 0,32,64,96
  const __bf16* gK = Kbh + (size_t)krr * HD + kcc;
  const __bf16* gV = Vbh + (size_t)vrr * SEQ + vcc;
  __bf16* sK = &lds_kp[krr * KSTR + kcc];
  __bf16* sV = &lds_v[vrr * VSTR + vcc];

  for (int kb = 0; kb < SEQ; kb += 128) {
    // ---- async stage K tile [128 keys][64 d], V tile [64 d][128 keys] ----
    {
      const __bf16* gk = gK + (size_t)kb * HD;
      copy16_async(sK,      gk);
      copy16_async(sK + 8,  gk + 8);
      copy16_async(sK + 16, gk + 16);
      copy16_async(sK + 24, gk + 24);
      const __bf16* gv = gV + kb;
      copy16_async(sV,      gv);
      copy16_async(sV + 8,  gv + 8);
      copy16_async(sV + 16, gv + 16);
      copy16_async(sV + 24, gv + 24);
    }
    wait_async_all();
    __syncthreads();

    // ---- scores S = Q K^T for 16 queries x 128 keys: 8 tiles x 2 WMMA ----
    v8f s[8];
#pragma unroll
    for (int nt = 0; nt < 8; ++nt) {
      const __bf16* kp = &lds_kp[(nt * 16 + l16) * KSTR + c0];
      v16bf b0 = SHUF16(*(const v8bf*)kp,        *(const v8bf*)(kp + 16));
      v16bf b1 = SHUF16(*(const v8bf*)(kp + 32), *(const v8bf*)(kp + 48));
      v8f z = v8f_zero();
      z = wmma_bf16(qa0, b0, z);
      z = wmma_bf16(qa1, b1, z);
      s[nt] = z;
    }
    __syncthreads();   // all waves done reading K-tile; P may reuse lds_kp

    // ---- online softmax (rows = lhalf*8 + j, duplicated across 16 lanes) ----
    float mloc[8];
#pragma unroll
    for (int j = 0; j < 8; ++j) {
      float mv = s[0][j];
#pragma unroll
      for (int nt = 1; nt < 8; ++nt) mv = fmaxf(mv, s[nt][j]);
      mloc[j] = mv;
    }
#pragma unroll
    for (int off = 1; off < 16; off <<= 1)
#pragma unroll
      for (int j = 0; j < 8; ++j)
        mloc[j] = fmaxf(mloc[j], __shfl_xor(mloc[j], off, 32));

    float mnew[8], alpha[8], rsum[8];
#pragma unroll
    for (int j = 0; j < 8; ++j) {
      mnew[j]  = fmaxf(m_i[j], mloc[j] * SCALE);
      alpha[j] = __expf(m_i[j] - mnew[j]);
      rsum[j]  = 0.0f;
    }
#pragma unroll
    for (int nt = 0; nt < 8; ++nt) {
#pragma unroll
      for (int j = 0; j < 8; ++j) {
        float pv = __expf(s[nt][j] * SCALE - mnew[j]);
        rsum[j] += pv;
        my_p[(lhalf * 8 + j) * VSTR + nt * 16 + l16] = f2bf(pv);
      }
    }
#pragma unroll
    for (int off = 1; off < 16; off <<= 1)
#pragma unroll
      for (int j = 0; j < 8; ++j)
        rsum[j] += __shfl_xor(rsum[j], off, 32);
#pragma unroll
    for (int j = 0; j < 8; ++j) {
      l_i[j] = l_i[j] * alpha[j] + rsum[j];
      m_i[j] = mnew[j];
    }
#pragma unroll
    for (int t = 0; t < 4; ++t)
#pragma unroll
      for (int j = 0; j < 8; ++j) o[t][j] *= alpha[j];

    // ---- O += P V : 4 K-chunks x 4 output tiles ----
#pragma unroll
    for (int kt = 0; kt < 4; ++kt) {
      const __bf16* pp = &my_p[l16 * VSTR + kt * 32 + c0];
      v16bf pa = SHUF16(*(const v8bf*)pp, *(const v8bf*)(pp + 16));
#pragma unroll
      for (int ot = 0; ot < 4; ++ot) {
        const __bf16* vp = &lds_v[(ot * 16 + l16) * VSTR + kt * 32 + c0];
        v16bf vb = SHUF16(*(const v8bf*)vp, *(const v8bf*)(vp + 16));
        o[ot] = wmma_bf16(pa, vb, o[ot]);
      }
    }
    __syncthreads();   // before restaging K/V tiles
  }

  // ---- normalize and scatter into AO[B*T][EMBED] (bf16) ----
  float inv[8];
#pragma unroll
  for (int j = 0; j < 8; ++j) inv[j] = 1.0f / l_i[j];
  const int h = bh % HEADS;
  const int b = bh / HEADS;
#pragma unroll
  for (int ot = 0; ot < 4; ++ot) {
    const int col = h * HD + ot * 16 + l16;
#pragma unroll
    for (int j = 0; j < 8; ++j) {
      const int q = q0 + lhalf * 8 + j;
      AO[(size_t)(b * SEQ + q) * EMBED + col] = f2bf(o[ot][j] * inv[j]);
    }
  }
}

// ---------------------------------------------------------------------------
// Host launcher
// ---------------------------------------------------------------------------
extern "C" void kernel_launch(void* const* d_in, const int* in_sizes, int n_in,
                              void* d_out, int out_size, void* d_ws, size_t ws_size,
                              hipStream_t stream) {
  (void)in_sizes; (void)n_in; (void)out_size; (void)ws_size;
  const float* x      = (const float*)d_in[0];
  const float* w_qkv  = (const float*)d_in[1];
  const float* b_qkv  = (const float*)d_in[2];
  const float* w_proj = (const float*)d_in[3];
  const float* b_proj = (const float*)d_in[4];
  float* out = (float*)d_out;

  char* ws = (char*)d_ws;
  size_t off = 0;
  auto take = [&](size_t bytes) -> char* {
    char* p = ws + off;
    off += (bytes + 255) & ~(size_t)255;
    return p;
  };
  __bf16* xb     = (__bf16*)take((size_t)MTOK * EMBED * 2);   // 12.6 MB
  __bf16* wqkvT  = (__bf16*)take((size_t)NQKV * EMBED * 2);   //  3.5 MB
  __bf16* wprojT = (__bf16*)take((size_t)EMBED * EMBED * 2);  //  1.2 MB
  __bf16* Qb     = (__bf16*)take((size_t)MTOK * EMBED * 2);   // 12.6 MB
  __bf16* Kb     = (__bf16*)take((size_t)MTOK * EMBED * 2);   // 12.6 MB
  __bf16* Vtb    = (__bf16*)take((size_t)MTOK * EMBED * 2);   // 12.6 MB
  __bf16* AO     = (__bf16*)take((size_t)MTOK * EMBED * 2);   // 12.6 MB

  // 1) precision conversion / weight transposes
  {
    int n = MTOK * EMBED;
    cvt_bf16_kernel<<<(n + 255) / 256, 256, 0, stream>>>(x, xb, n);
  }
  {
    int n = EMBED * NQKV;
    cvt_transpose_kernel<<<(n + 255) / 256, 256, 0, stream>>>(w_qkv, wqkvT, EMBED, NQKV);
  }
  {
    int n = EMBED * EMBED;
    cvt_transpose_kernel<<<(n + 255) / 256, 256, 0, stream>>>(w_proj, wprojT, EMBED, EMBED);
  }

  // 2) QKV projection GEMM -> Q, K, Vt (bf16)
  {
    dim3 grid(MTOK / BM, NQKV / BN);   // 64 x 18
    gemm_kernel<0><<<grid, 256, 0, stream>>>(xb, wqkvT, b_qkv, Qb, Kb, Vtb, nullptr);
  }

  // 3) fused attention -> AO (bf16)
  {
    dim3 grid(BATCH * HEADS, SEQ / 128);  // 48 x 16
    attn_kernel<<<grid, 256, 0, stream>>>(Qb, Kb, Vtb, AO);
  }

  // 4) output projection GEMM -> out (f32)
  {
    dim3 grid(MTOK / BM, EMBED / BN);  // 64 x 6
    gemm_kernel<1><<<grid, 256, 0, stream>>>(AO, wprojT, b_proj, nullptr, nullptr, nullptr, out);
  }
}
